// Attention_77429670412575
// MI455X (gfx1250) — matrix operand order repaired and checked by reading, not compiled, not run on hardware
//
#include <hip/hip_runtime.h>
#include <hip/hip_bf16.h>
#include <math.h>

// Problem constants (reference: B=16, T=1024, F=512)
#define B_   16
#define T_   1024
#define F_   512
#define M_   (B_ * T_)          // 16384 rows of the fused GEMM
#define EPS_ 1e-7f

#define NSEG_   8               // prefix-sum segments over T
#define SEGLEN_ (T_ / NSEG_)    // 128
#define NFCH_   4               // F chunks of 128
#define FCH_    (F_ / NFCH_)    // 128

#define NBLK_   4               // N-tiles register-blocked per K pass (64 cols)

typedef __attribute__((ext_vector_type(16))) __bf16 v16bf;
typedef __attribute__((ext_vector_type(8)))  __bf16 v8bf;
typedef __attribute__((ext_vector_type(8)))  float  v8f;

// ---------------------------------------------------------------------------
// 1) fp32 -> (bf16 hi, bf16 lo) split of x.  hi is RNE-rounded; lo is the
//    residual, so hi*hi + hi*lo + lo*hi recovers ~fp32 products.
// ---------------------------------------------------------------------------
__global__ __launch_bounds__(256)
void split_x_kernel(const float* __restrict__ x,
                    __bf16* __restrict__ xhi, __bf16* __restrict__ xlo, int n) {
  int i = blockIdx.x * blockDim.x + threadIdx.x;
  if (i < n) {
    float f = x[i];
    __bf16 h = (__bf16)f;
    xhi[i] = h;
    xlo[i] = (__bf16)(f - (float)h);
  }
}

// ---------------------------------------------------------------------------
// 2) W (F x F, row-major f,g) -> transposed split Wt[g][f] so that the WMMA
//    B-fragment (32x16, K x N) is a single contiguous 32-byte load per lane:
//    lane n (n = lane&15) needs W[k][n0+n] for 16 consecutive k = Wt[n0+n][k..].
// ---------------------------------------------------------------------------
__global__ __launch_bounds__(256)
void transpose_split_W_kernel(const float* __restrict__ W,
                              __bf16* __restrict__ wthi, __bf16* __restrict__ wtlo) {
  int i = blockIdx.x * blockDim.x + threadIdx.x;   // i = g*F + f
  if (i < F_ * F_) {
    int g = i / F_, f = i - g * F_;
    float v = W[f * F_ + g];
    __bf16 h = (__bf16)v;
    wthi[i] = h;
    wtlo[i] = (__bf16)(v - (float)h);
  }
}

// ---------------------------------------------------------------------------
// 3) Fused alpha kernel:  w[row] = exp( u . tanh(x[row]*W + bias) ) * mask[row]
//    One wave per 16-row M-tile.  NBLK_=4 N-tiles are register-blocked per
//    K pass so each A-fragment load feeds 12 WMMAs (~0.67KB L0 traffic/WMMA).
//    v_att (B,T,F) never touches memory.
// ---------------------------------------------------------------------------
__global__ __launch_bounds__(256)
void alpha_kernel(const __bf16* __restrict__ xhi, const __bf16* __restrict__ xlo,
                  const __bf16* __restrict__ wthi, const __bf16* __restrict__ wtlo,
                  const float* __restrict__ bias, const float* __restrict__ u,
                  const unsigned char* __restrict__ mask,
                  float* __restrict__ wout) {
  const int wid  = threadIdx.x >> 5;        // wave in block (8 waves)
  const int lane = threadIdx.x & 31;        // wave32
  const int half = lane >> 4;               // 0: lanes 0-15, 1: lanes 16-31
  const int lid  = lane & 15;
  const int tile = blockIdx.x * 8 + wid;    // 1024 tiles of 16 rows
  const int arow = tile * 16 + lid;         // A-fragment row for this lane

  const __bf16* arh = xhi + (size_t)arow * F_;
  const __bf16* arl = xlo + (size_t)arow * F_;

  float partial[8];
#pragma unroll
  for (int r = 0; r < 8; ++r) partial[r] = 0.f;

  for (int n0 = 0; n0 < F_; n0 += 16 * NBLK_) {     // 8 outer N chunks
    v8f acc[NBLK_];
#pragma unroll
    for (int j = 0; j < NBLK_; ++j) acc[j] = (v8f){};

    // Per-lane B row pointers for the 4 N-tiles of this chunk
    const __bf16* brh[NBLK_];
    const __bf16* brl[NBLK_];
#pragma unroll
    for (int j = 0; j < NBLK_; ++j) {
      brh[j] = wthi + (size_t)(n0 + 16 * j + lid) * F_ + half * 16;
      brl[j] = wtlo + (size_t)(n0 + 16 * j + lid) * F_ + half * 16;
    }

#pragma unroll 2
    for (int k0 = 0; k0 < F_; k0 += 32) {
      // A 16x32 bf16 layout: lane(lid,half=0): K=k0..k0+15
      //                      lane(lid,half=1): K=k0+16..k0+31
      v8bf a0h = *(const v8bf*)(arh + k0 + half * 16);
      v8bf a1h = *(const v8bf*)(arh + k0 + 8 + half * 16);
      v8bf a0l = *(const v8bf*)(arl + k0 + half * 16);
      v8bf a1l = *(const v8bf*)(arl + k0 + 8 + half * 16);
      v16bf ah = __builtin_shufflevector(a0h, a1h, 0,1,2,3,4,5,6,7,8,9,10,11,12,13,14,15);
      v16bf al = __builtin_shufflevector(a0l, a1l, 0,1,2,3,4,5,6,7,8,9,10,11,12,13,14,15);
#pragma unroll
      for (int j = 0; j < NBLK_; ++j) {
        // B 32x16 bf16: lane holds N, 16 consecutive K (contiguous 32B)
        v16bf bh = *(const v16bf*)(brh[j] + k0);
        v16bf bl = *(const v16bf*)(brl[j] + k0);
        // Split-bf16 fp32 emulation: hi*hi + hi*lo + lo*hi
        acc[j] = __builtin_amdgcn_wmma_f32_16x16x32_bf16(false, ah, false, bh, (short)0, acc[j], false, false);
        acc[j] = __builtin_amdgcn_wmma_f32_16x16x32_bf16(false, ah, false, bl, (short)0, acc[j], false, false);
        acc[j] = __builtin_amdgcn_wmma_f32_16x16x32_bf16(false, al, false, bh, (short)0, acc[j], false, false);
      }
    }

    // C layout: VGPR r, lanes 0-15 -> M=r, lanes 16-31 -> M=8+r; N = lane&15
#pragma unroll
    for (int j = 0; j < NBLK_; ++j) {
      const int   col = n0 + 16 * j + lid;
      const float bv  = bias[col];
      const float uv  = u[col];
#pragma unroll
      for (int r = 0; r < 8; ++r)
        partial[r] += tanhf(acc[j][r] + bv) * uv;
    }
  }

  // Reduce over the 16 N-lanes of each half-wave (bit 4 untouched).
#pragma unroll
  for (int r = 0; r < 8; ++r) {
    float v = partial[r];
    v += __shfl_xor(v, 1, 32);
    v += __shfl_xor(v, 2, 32);
    v += __shfl_xor(v, 4, 32);
    v += __shfl_xor(v, 8, 32);
    partial[r] = v;
  }
  if (lid == 0) {
#pragma unroll
    for (int r = 0; r < 8; ++r) {
      int orow = tile * 16 + half * 8 + r;       // == b*T + t
      float m  = mask[orow] ? 1.f : 0.f;
      wout[orow] = expf(partial[r]) * m;         // e = exp(alpha)*mask
    }
  }
}

// ---------------------------------------------------------------------------
// 4) Per-batch inclusive scan of w -> rinv[b,t] = 1 / (cumsum(w) + eps)
// ---------------------------------------------------------------------------
__global__ __launch_bounds__(32)
void scan_kernel(const float* __restrict__ w, float* __restrict__ rinv) {
  int b = blockIdx.x;
  int lane = threadIdx.x;                        // one wave32
  float carry = 0.f;
  for (int t0 = 0; t0 < T_; t0 += 32) {
    float v = w[b * T_ + t0 + lane];
#pragma unroll
    for (int off = 1; off < 32; off <<= 1) {
      float n = __shfl_up(v, off, 32);
      if (lane >= off) v += n;
    }
    float den = carry + v;
    rinv[b * T_ + t0 + lane] = 1.f / (den + EPS_);
    carry += __shfl(v, 31, 32);
  }
}

// ---------------------------------------------------------------------------
// 5a) Segment sums of w*x over T segments (parallelizes the prefix in t).
// ---------------------------------------------------------------------------
__global__ __launch_bounds__(FCH_)
void segsum_kernel(const float* __restrict__ x, const float* __restrict__ w,
                   float* __restrict__ segsum) {
  int id  = blockIdx.x;                          // B*NSEG*NFCH = 512 blocks
  int fc  = id & (NFCH_ - 1);
  int seg = (id >> 2) & (NSEG_ - 1);
  int b   = id >> 5;
  int f   = fc * FCH_ + threadIdx.x;
  int t0  = seg * SEGLEN_;
  float acc = 0.f;
  for (int t = t0; t < t0 + SEGLEN_; ++t)
    acc += w[b * T_ + t] * x[((size_t)(b * T_ + t)) * F_ + f];
  segsum[(b * NSEG_ + seg) * F_ + f] = acc;
}

// ---------------------------------------------------------------------------
// 5b) Within-segment running prefix (seeded by prior segment sums), scaled by
//     the softmax reciprocal:  out[b,t,f] = prefix(w*x) * rinv[b,t]
// ---------------------------------------------------------------------------
__global__ __launch_bounds__(FCH_)
void prefix_out_kernel(const float* __restrict__ x, const float* __restrict__ w,
                       const float* __restrict__ rinv, const float* __restrict__ segsum,
                       float* __restrict__ out) {
  int id  = blockIdx.x;
  int fc  = id & (NFCH_ - 1);
  int seg = (id >> 2) & (NSEG_ - 1);
  int b   = id >> 5;
  int f   = fc * FCH_ + threadIdx.x;
  float acc = 0.f;
  for (int s = 0; s < seg; ++s)
    acc += segsum[(b * NSEG_ + s) * F_ + f];
  int t0 = seg * SEGLEN_;
  for (int t = t0; t < t0 + SEGLEN_; ++t) {
    size_t xi = ((size_t)(b * T_ + t)) * F_ + f;
    acc += w[b * T_ + t] * x[xi];
    out[xi] = acc * rinv[b * T_ + t];
  }
}

// ---------------------------------------------------------------------------
// Host launcher
// ---------------------------------------------------------------------------
extern "C" void kernel_launch(void* const* d_in, const int* in_sizes, int n_in,
                              void* d_out, int out_size, void* d_ws, size_t ws_size,
                              hipStream_t stream) {
  (void)in_sizes; (void)n_in; (void)out_size; (void)ws_size;
  const float*         x    = (const float*)d_in[0];
  const unsigned char* mask = (const unsigned char*)d_in[1];  // jnp bool -> 1 byte
  const float*         W    = (const float*)d_in[2];
  const float*         bias = (const float*)d_in[3];
  const float*         u    = (const float*)d_in[4];
  float*               out  = (float*)d_out;

  // Workspace layout (~34.5 MB total, 256B-aligned blocks)
  char* ws = (char*)d_ws;
  size_t off = 0;
  auto take = [&](size_t bytes) -> void* {
    void* p = ws + off;
    off += (bytes + 255) & ~(size_t)255;
    return p;
  };
  __bf16* xhi    = (__bf16*)take((size_t)M_ * F_ * sizeof(__bf16));  // 16 MB
  __bf16* xlo    = (__bf16*)take((size_t)M_ * F_ * sizeof(__bf16));  // 16 MB
  __bf16* wthi   = (__bf16*)take((size_t)F_ * F_ * sizeof(__bf16));  // 512 KB
  __bf16* wtlo   = (__bf16*)take((size_t)F_ * F_ * sizeof(__bf16));  // 512 KB
  float*  wbuf   = (float*)take((size_t)M_ * sizeof(float));         // 64 KB
  float*  rbuf   = (float*)take((size_t)M_ * sizeof(float));         // 64 KB
  float*  segbuf = (float*)take((size_t)B_ * NSEG_ * F_ * sizeof(float)); // 256 KB

  const int nX = M_ * F_;
  split_x_kernel<<<(nX + 255) / 256, 256, 0, stream>>>(x, xhi, xlo, nX);
  transpose_split_W_kernel<<<(F_ * F_ + 255) / 256, 256, 0, stream>>>(W, wthi, wtlo);

  // 1024 M-tiles, 8 waves (tiles) per 256-thread block -> 128 blocks
  alpha_kernel<<<M_ / (16 * 8), 256, 0, stream>>>(xhi, xlo, wthi, wtlo,
                                                  bias, u, mask, wbuf);

  scan_kernel<<<B_, 32, 0, stream>>>(wbuf, rbuf);

  segsum_kernel<<<B_ * NSEG_ * NFCH_, FCH_, 0, stream>>>(x, wbuf, segbuf);
  prefix_out_kernel<<<B_ * NSEG_ * NFCH_, FCH_, 0, stream>>>(x, wbuf, rbuf,
                                                             segbuf, out);
}
